// AttentionLayer_67035849556293
// MI455X (gfx1250) — compile-verified, matrix operand order checked by reading, not run on hardware
//
#include <hip/hip_runtime.h>
#include <hip/hip_bf16.h>
#include <math.h>

// ---------------------------------------------------------------------------
// CDNA5 / gfx1250 attention layer.
//   B=4, Q=K=4096, D=IN=256.
//   GEMMs on v_wmma_f32_16x16x32_bf16 (bf16 operands, f32 accum).
//   Flash attention with double-buffered GLOBAL_LOAD_ASYNC_TO_LDS staging of
//   K/V chunks (ASYNCcnt pipeline) shared by all 8 waves of the block.
// ---------------------------------------------------------------------------

typedef __attribute__((ext_vector_type(16))) __bf16 v16bf;
typedef __attribute__((ext_vector_type(8)))  float  v8f;

#define WMMA_BF16(a, b, c) \
    __builtin_amdgcn_wmma_f32_16x16x32_bf16(false, (a), false, (b), (short)0, (c), false, false)

// LDS byte offset of a __shared__ object: for flat pointers in the LDS
// aperture the low 32 bits are the LDS address (ISA 10.2 aperture rules).
#define LDS_OFF(p) ((uint32_t)(uintptr_t)(p))

// ---------------------------------------------------------------------------
// Fragment loaders (wave32 layouts per CDNA5 ISA 7.12.2).
// 16-bit A 16x32: lane L holds row M=L&15; halves 0..7 -> K=kb..kb+7,
// halves 8..15 -> K=16+kb.., kb=(L>=16)*8. B 32x16 mirrors this as rows of
// B^T, so any row-major [N,K] matrix loads directly as a B fragment.
// ---------------------------------------------------------------------------
union Frag16 { uint4 u[2]; v16bf v; };

__device__ __forceinline__ v16bf frag_from_bf16(const __bf16* __restrict__ base,
                                                int ld, int lane) {
    const int row = lane & 15;
    const int kb  = (lane >> 4) << 3;          // 0 or 8
    const __bf16* p = base + (size_t)row * ld + kb;
    Frag16 f;
    f.u[0] = *(const uint4*)(p);
    f.u[1] = *(const uint4*)(p + 16);
    return f.v;
}

__device__ __forceinline__ v16bf frag_from_f32(const float* __restrict__ base,
                                               int ld, int lane) {
    const int row = lane & 15;
    const int kb  = (lane >> 4) << 3;
    const float* p = base + (size_t)row * ld + kb;
    v16bf a;
#pragma unroll
    for (int j = 0; j < 8; ++j) {
        a[j]     = (__bf16)p[j];
        a[j + 8] = (__bf16)p[16 + j];
    }
    return a;
}

// ---------------------------------------------------------------------------
// fp32 -> bf16 weight conversion
// ---------------------------------------------------------------------------
__global__ void cvt_w_kernel(const float* __restrict__ src,
                             __bf16* __restrict__ dst, int n) {
    int i = blockIdx.x * blockDim.x + threadIdx.x;
    if (i * 4 + 3 < n) {
        float4 f = ((const float4*)src)[i];
        dst[i * 4 + 0] = (__bf16)f.x;
        dst[i * 4 + 1] = (__bf16)f.y;
        dst[i * 4 + 2] = (__bf16)f.z;
        dst[i * 4 + 3] = (__bf16)f.w;
    }
}

// ---------------------------------------------------------------------------
// Projection GEMM: Y[m,o] = sum_i X[m,i]*W[o,i] + bias[o]   (X fp32, W bf16)
// mode 0: Y row-major [M,256] (Qp,Kp). mode 1: per-batch transposed Vt[b][o][k].
// ---------------------------------------------------------------------------
__global__ void proj_kernel(const float* __restrict__ X,
                            const __bf16* __restrict__ W,
                            const float* __restrict__ bias,
                            __bf16* __restrict__ Y, int mode) {
    const int lane = threadIdx.x & 31;
    const int wave = threadIdx.x >> 5;
    const int m0 = blockIdx.x * 64 + wave * 16;
    const int n0 = blockIdx.y * 16;

    v8f acc = {0.f, 0.f, 0.f, 0.f, 0.f, 0.f, 0.f, 0.f};
#pragma unroll
    for (int kc = 0; kc < 256; kc += 32) {
        v16bf a = frag_from_f32(X + (size_t)m0 * 256 + kc, 256, lane);
        v16bf b = frag_from_bf16(W + (size_t)n0 * 256 + kc, 256, lane);
        acc = WMMA_BF16(a, b, acc);
    }
    const int n = n0 + (lane & 15);
    const int mbase = m0 + ((lane >> 4) << 3);
    const float bn = bias[n];
#pragma unroll
    for (int r = 0; r < 8; ++r) {
        float y = acc[r] + bn;
        int m = mbase + r;
        if (mode == 0) {
            Y[(size_t)m * 256 + n] = (__bf16)y;
        } else {
            int bi = m >> 12, kk = m & 4095;
            Y[((size_t)(bi * 256 + n) << 12) + kk] = (__bf16)y;
        }
    }
}

// ---------------------------------------------------------------------------
// Flash attention.
//  Block = 256 threads = 8 waves = 128 queries (one batch).
//  Each wave: 16 queries x full D=256 (16 f32 accum tiles).
//  K/V 64-key chunks are async-DMA'd into double-buffered LDS by the whole
//  block (GLOBAL_LOAD_ASYNC_TO_LDS_B128 / s_wait_asynccnt), consumed via
//  ds_load_b128 fragment reads by all waves.
//  LDS rows are padded (K:264, V:72, P:72 bf16) to spread banks.
// ---------------------------------------------------------------------------
__global__ __launch_bounds__(256, 1)
void flash_attn_kernel(const __bf16* __restrict__ Qp,   // [B*Q, 256] row-major
                       const __bf16* __restrict__ Kp,   // [B*K, 256] row-major
                       const __bf16* __restrict__ Vt,   // [B, 256, 4096]
                       const int*    __restrict__ mask, // [B*Q]
                       __bf16*       __restrict__ Att)  // [B*Q, 256]
{
    __shared__ __align__(16) __bf16 smemK[2][64][264];  //  67.6 KB (row = key)
    __shared__ __align__(16) __bf16 smemV[2][256][72];  //  73.7 KB (row = dim)
    __shared__ __align__(16) __bf16 pbuf[8][16][72];    //  18.4 KB per-wave P

    const int tid   = threadIdx.x;
    const int lane  = tid & 31;
    const int wave  = tid >> 5;
    const int q0    = (blockIdx.x * 8 + wave) * 16;     // global row in [0, B*Q)
    const int bidx  = (blockIdx.x * 128) >> 12;         // batch (128 q / block)
    const int rbase = (lane >> 4) << 3;                 // 0 / 8
    const int nl    = lane & 15;

    const __bf16* Kbase = Kp + ((size_t)bidx << 12) * 256;
    const __bf16* Vbase = Vt + (size_t)bidx * 256 * 4096;

    const uint32_t ldsK[2] = { LDS_OFF(&smemK[0][0][0]), LDS_OFF(&smemK[1][0][0]) };
    const uint32_t ldsV[2] = { LDS_OFF(&smemV[0][0][0]), LDS_OFF(&smemV[1][0][0]) };

    // -------- async chunk issue: 64 keys of K (64x512B) + V (256x128B) ------
    // 16 x b128 per thread-block-slice instruction group; 16 instrs per wave.
    auto issue_chunk = [&](int buf, int kc) {
        const __bf16* ksad = Kbase + (size_t)kc * 256;   // + kc*512 bytes
        const __bf16* vsad = Vbase + kc;                 // + kc*2   bytes
#pragma unroll
        for (int i = 0; i < 8; ++i) {
            uint32_t L    = (uint32_t)(i * 4096 + tid * 16);   // chunk byte
            uint32_t krow = L >> 9, kcol = L & 511;
            uint32_t kldsa = ldsK[buf] + krow * 528 + kcol;
            uint32_t kvoff = krow * 512 + kcol;
            asm volatile("global_load_async_to_lds_b128 %0, %1, %2"
                         :: "v"(kldsa), "v"(kvoff), "s"(ksad) : "memory");
            uint32_t vrow = L >> 7, vcol = L & 127;
            uint32_t vldsa = ldsV[buf] + vrow * 144 + vcol;
            uint32_t vvoff = vrow * 8192 + vcol;
            asm volatile("global_load_async_to_lds_b128 %0, %1, %2"
                         :: "v"(vldsa), "v"(vvoff), "s"(vsad) : "memory");
        }
    };

    // -------- hoisted Q fragments + per-row mask lengths --------------------
    v16bf aq[8];
#pragma unroll
    for (int dc = 0; dc < 8; ++dc)
        aq[dc] = frag_from_bf16(Qp + (size_t)q0 * 256 + dc * 32, 256, lane);

    int maskv[8];
#pragma unroll
    for (int r = 0; r < 8; ++r) maskv[r] = mask[q0 + rbase + r];

    v8f O[16];
#pragma unroll
    for (int t = 0; t < 16; ++t)
        O[t] = (v8f){0.f, 0.f, 0.f, 0.f, 0.f, 0.f, 0.f, 0.f};
    float mrow[8], lrow[8];
#pragma unroll
    for (int r = 0; r < 8; ++r) { mrow[r] = -3.0e38f; lrow[r] = 0.f; }

    issue_chunk(0, 0);                                   // prime the pipeline

    for (int c = 0; c < 64; ++c) {
        const int kc  = c * 64;
        const int cur = c & 1;
        if (c + 1 < 64) {
            issue_chunk(cur ^ 1, kc + 64);               // overlap next chunk
            asm volatile("s_wait_asynccnt 0x10" ::: "memory");  // cur landed
        } else {
            asm volatile("s_wait_asynccnt 0x0" ::: "memory");
        }
        __syncthreads();                                 // all waves' DMA done

        // ---- S = Q K^T for 16q x 64k (4 tiles), K in LDS ----
        v8f S[4];
#pragma unroll
        for (int t = 0; t < 4; ++t)
            S[t] = (v8f){0.f, 0.f, 0.f, 0.f, 0.f, 0.f, 0.f, 0.f};
#pragma unroll
        for (int dc = 0; dc < 8; ++dc) {
#pragma unroll
            for (int t = 0; t < 4; ++t) {
                v16bf bk = frag_from_bf16(&smemK[cur][t * 16][dc * 32], 264, lane);
                S[t] = WMMA_BF16(aq[dc], bk, S[t]);
            }
        }
        // ---- mask + scale (masked_fill(-1e9) THEN /16 -> -6.25e7) ----
#pragma unroll
        for (int t = 0; t < 4; ++t) {
            int kidx = kc + t * 16 + nl;
#pragma unroll
            for (int r = 0; r < 8; ++r) {
                float s = S[t][r];
                S[t][r] = (kidx < maskv[r]) ? (s * 0.0625f) : -6.25e7f;
            }
        }
        // ---- online softmax (row m = one VGPR across a 16-lane half) ----
        float corr[8];
#pragma unroll
        for (int r = 0; r < 8; ++r) {
            float mc = fmaxf(fmaxf(S[0][r], S[1][r]), fmaxf(S[2][r], S[3][r]));
            mc = fmaxf(mc, __shfl_xor(mc, 1));
            mc = fmaxf(mc, __shfl_xor(mc, 2));
            mc = fmaxf(mc, __shfl_xor(mc, 4));
            mc = fmaxf(mc, __shfl_xor(mc, 8));
            float mnew = fmaxf(mrow[r], mc);
            corr[r] = __expf(mrow[r] - mnew);
            mrow[r] = mnew;
            float ls = 0.f;
#pragma unroll
            for (int t = 0; t < 4; ++t) {
                float p = __expf(S[t][r] - mnew);
                S[t][r] = p;
                ls += p;
            }
            ls += __shfl_xor(ls, 1);
            ls += __shfl_xor(ls, 2);
            ls += __shfl_xor(ls, 4);
            ls += __shfl_xor(ls, 8);
            lrow[r] = lrow[r] * corr[r] + ls;
        }
#pragma unroll
        for (int dt = 0; dt < 16; ++dt)
#pragma unroll
            for (int r = 0; r < 8; ++r) O[dt][r] *= corr[r];

        // ---- stage P (bf16) via per-wave LDS, reread as A fragments ----
#pragma unroll
        for (int t = 0; t < 4; ++t)
#pragma unroll
            for (int r = 0; r < 8; ++r)
                pbuf[wave][rbase + r][t * 16 + nl] = (__bf16)S[t][r];
        asm volatile("s_wait_dscnt 0x0" ::: "memory");   // same-wave LDS RAW
        v16bf p0 = frag_from_bf16(&pbuf[wave][0][0],  72, lane);
        v16bf p1 = frag_from_bf16(&pbuf[wave][0][32], 72, lane);

        // ---- O += P @ V, V^T rows streamed from LDS ----
#pragma unroll
        for (int dt = 0; dt < 16; ++dt) {
            v16bf bv0 = frag_from_bf16(&smemV[cur][dt * 16][0],  72, lane);
            v16bf bv1 = frag_from_bf16(&smemV[cur][dt * 16][32], 72, lane);
            O[dt] = WMMA_BF16(p0, bv0, O[dt]);
            O[dt] = WMMA_BF16(p1, bv1, O[dt]);
        }
        __syncthreads();          // buffer (cur) safe to overwrite at c+2
    }

    // ---- finalize: O /= l, write Att (bf16) ----
    float inv[8];
#pragma unroll
    for (int r = 0; r < 8; ++r) inv[r] = 1.0f / lrow[r];
#pragma unroll
    for (int dt = 0; dt < 16; ++dt)
#pragma unroll
        for (int r = 0; r < 8; ++r) {
            float y = O[dt][r] * inv[r];
            Att[(size_t)(q0 + rbase + r) * 256 + dt * 16 + nl] = (__bf16)y;
        }
}

// ---------------------------------------------------------------------------
// Output projection: out[m,o] = sum_d Att[m,d]*Wo[o,d] + bo[o]  (fp32 out)
// ---------------------------------------------------------------------------
__global__ void oproj_kernel(const __bf16* __restrict__ A,
                             const __bf16* __restrict__ W,
                             const float* __restrict__ bias,
                             float* __restrict__ out) {
    const int lane = threadIdx.x & 31;
    const int wave = threadIdx.x >> 5;
    const int m0 = blockIdx.x * 64 + wave * 16;
    const int n0 = blockIdx.y * 16;

    v8f acc = {0.f, 0.f, 0.f, 0.f, 0.f, 0.f, 0.f, 0.f};
#pragma unroll
    for (int kc = 0; kc < 256; kc += 32) {
        v16bf a = frag_from_bf16(A + (size_t)m0 * 256 + kc, 256, lane);
        v16bf b = frag_from_bf16(W + (size_t)n0 * 256 + kc, 256, lane);
        acc = WMMA_BF16(a, b, acc);
    }
    const int n = n0 + (lane & 15);
    const int mbase = m0 + ((lane >> 4) << 3);
    const float bn = bias[n];
#pragma unroll
    for (int r = 0; r < 8; ++r)
        out[(size_t)(mbase + r) * 256 + n] = acc[r] + bn;
}

// ---------------------------------------------------------------------------
// Host-side launch. Workspace layout (bytes):
//   [0, 512K)   : Wq/Wk/Wv/Wo bf16 (128K each)
//   then 8 MiB each: Qp, Kp, Vt, Att (bf16). Total ~34 MB.
// ---------------------------------------------------------------------------
extern "C" void kernel_launch(void* const* d_in, const int* in_sizes, int n_in,
                              void* d_out, int out_size, void* d_ws, size_t ws_size,
                              hipStream_t stream) {
    const float* qry = (const float*)d_in[0];
    const float* key = (const float*)d_in[1];
    const float* val = (const float*)d_in[2];
    const int*   msk = (const int*)d_in[3];
    const float* Wq  = (const float*)d_in[4];
    const float* bq  = (const float*)d_in[5];
    const float* Wk  = (const float*)d_in[6];
    const float* bk  = (const float*)d_in[7];
    const float* Wv  = (const float*)d_in[8];
    const float* bv  = (const float*)d_in[9];
    const float* Wo  = (const float*)d_in[10];
    const float* bo  = (const float*)d_in[11];

    char* ws = (char*)d_ws;
    const size_t WSZ = 1u << 17;
    const size_t MSZ = 8u << 20;
    __bf16* Wq_b = (__bf16*)(ws + 0 * WSZ);
    __bf16* Wk_b = (__bf16*)(ws + 1 * WSZ);
    __bf16* Wv_b = (__bf16*)(ws + 2 * WSZ);
    __bf16* Wo_b = (__bf16*)(ws + 3 * WSZ);
    __bf16* Qp   = (__bf16*)(ws + 4 * WSZ + 0 * MSZ);
    __bf16* Kp   = (__bf16*)(ws + 4 * WSZ + 1 * MSZ);
    __bf16* Vt   = (__bf16*)(ws + 4 * WSZ + 2 * MSZ);
    __bf16* Att  = (__bf16*)(ws + 4 * WSZ + 3 * MSZ);

    cvt_w_kernel<<<64, 256, 0, stream>>>(Wq, Wq_b, 256 * 256);
    cvt_w_kernel<<<64, 256, 0, stream>>>(Wk, Wk_b, 256 * 256);
    cvt_w_kernel<<<64, 256, 0, stream>>>(Wv, Wv_b, 256 * 256);
    cvt_w_kernel<<<64, 256, 0, stream>>>(Wo, Wo_b, 256 * 256);

    dim3 pg(256, 16);
    proj_kernel<<<pg, 128, 0, stream>>>(qry, Wq_b, bq, Qp, 0);
    proj_kernel<<<pg, 128, 0, stream>>>(key, Wk_b, bk, Kp, 0);
    proj_kernel<<<pg, 128, 0, stream>>>(val, Wv_b, bv, Vt, 1);

    flash_attn_kernel<<<128, 256, 0, stream>>>(Qp, Kp, Vt, msk, Att);

    oproj_kernel<<<pg, 128, 0, stream>>>(Att, Wo_b, bo, (float*)d_out);
}
